// Model_7516192768052
// MI455X (gfx1250) — compile-verified, matrix operand order checked by reading, not run on hardware
//
#include <hip/hip_runtime.h>
#include <hip/hip_bf16.h>

typedef __bf16 bf16;
typedef __attribute__((ext_vector_type(16))) __bf16 v16bf;
typedef __attribute__((ext_vector_type(8)))  __bf16 v8bf;
typedef __attribute__((ext_vector_type(4)))  __bf16 v4bf;
typedef __attribute__((ext_vector_type(8)))  float  v8f;

#define SEQ   768
#define PRED  192
#define KTOT  32768   // 256 i * 64 modes * 2 (re/im)
#define LDST  40      // LDS row stride (bf16): 80B = 5*16B, 16B aligned for b128

__device__ __forceinline__ unsigned lane_id() {
  return __builtin_amdgcn_mbcnt_hi(~0u, __builtin_amdgcn_mbcnt_lo(~0u, 0u));
}
__device__ __forceinline__ unsigned short f2bf(float f) {
  union { float f; unsigned u; } v; v.f = f;
  unsigned r = v.u + 0x7fffu + ((v.u >> 16) & 1u);
  return (unsigned short)(r >> 16);
}
__device__ __forceinline__ float bf2f(unsigned short b) {
  union { unsigned u; float f; } v; v.u = ((unsigned)b) << 16;
  return v.f;
}
__device__ __forceinline__ void splitf(float x, unsigned short& h, unsigned short& l) {
  h = f2bf(x);
  l = f2bf(x - bf2f(h));
}
// fragment = concat(row[8h..8h+8), row[16+8h..16+8h+8)) -> two ds_load_b128
__device__ __forceinline__ v16bf frag_load(const bf16* row, int h) {
  v8bf lo = *(const v8bf*)(row + 8 * h);
  v8bf hi = *(const v8bf*)(row + 16 + 8 * h);
  return __builtin_shufflevector(lo, hi, 0, 1, 2, 3, 4, 5, 6, 7,
                                 8, 9, 10, 11, 12, 13, 14, 15);
}

// ---------------- normalization: xn[bj][t], mean/std per bj ----------------
__global__ void k_normalize(const float* __restrict__ x, const float* __restrict__ aw,
                            const float* __restrict__ ab, float* __restrict__ xn,
                            float* __restrict__ meanstd) {
  __shared__ float s1[128], s2[128];
  __shared__ float ms[2];
  int bj = blockIdx.x, b = bj >> 6, c = bj & 63, t = threadIdx.x;
  float sum = 0.f, sq = 0.f;
  for (int q = 0; q < 6; ++q) {
    float v = x[((size_t)b * SEQ + (t + 128 * q)) * 64 + c];
    sum += v; sq += v * v;
  }
  s1[t] = sum; s2[t] = sq; __syncthreads();
  for (int w = 64; w > 0; w >>= 1) {
    if (t < w) { s1[t] += s1[t + w]; s2[t] += s2[t + w]; }
    __syncthreads();
  }
  if (t == 0) {
    float mean = s1[0] / (float)SEQ;
    float var  = s2[0] / (float)SEQ - mean * mean;
    float sd   = sqrtf(var + 1e-5f);
    ms[0] = mean; ms[1] = sd;
    meanstd[bj] = mean; meanstd[256 + bj] = sd;
  }
  __syncthreads();
  float mean = ms[0], sd = ms[1], w_ = aw[c], b_ = ab[c];
  for (int q = 0; q < 6; ++q) {
    int tt = t + 128 * q;
    float v = x[((size_t)b * SEQ + tt) * 64 + c];
    xn[(size_t)bj * SEQ + tt] = (v - mean) / sd * w_ + b_;
  }
}

// ---------------- DFT twiddle tables E_s[d][m] = exp(-2*pi*i*d*m/L) --------
__global__ void k_etab(float* __restrict__ E) {
  const int Ls[3]   = {192, 384, 768};
  const int Eoff[3] = {0, 24576, 73728};
  int s = blockIdx.y;
  int idx = blockIdx.x * 256 + threadIdx.x;
  int L = Ls[s];
  int d = idx >> 6, m = idx & 63;
  if (d >= L) return;
  int r = (d * m) % L;  // exact range reduction
  float ang = -6.2831853071795864769f * (float)r / (float)L;
  float* e = E + Eoff[s] + (size_t)idx * 2;
  e[0] = cosf(ang); e[1] = sinf(ang);
}

// ---------------- v_k = A^k B, rows 0..63 sequentially ---------------------
__global__ void k_vseed(const float* __restrict__ A, const float* __restrict__ Bv,
                        float* __restrict__ VF, unsigned short* __restrict__ VH,
                        unsigned short* __restrict__ VL) {
  const int Voff[3] = {0, 192, 576};
  int s = blockIdx.x, n = threadIdx.x;
  __shared__ float vb[2][256];
  float* vf = VF + (size_t)Voff[s] * 256;
  unsigned short* vh = VH + (size_t)Voff[s] * 256;
  unsigned short* vl = VL + (size_t)Voff[s] * 256;
  float v0 = Bv[s * 256 + n];
  vb[0][n] = v0; vf[n] = v0;
  splitf(v0, vh[n], vl[n]);
  __syncthreads();
  const float4* a4 = (const float4*)(A + (size_t)s * 65536 + (size_t)n * 256);
  int cur = 0;
  for (int k = 1; k < 64; ++k) {
    float acc = 0.f;
    const float* vc = vb[cur];
    for (int j = 0; j < 64; ++j) {
      float4 a = a4[j];
      acc += a.x * vc[4*j] + a.y * vc[4*j+1] + a.z * vc[4*j+2] + a.w * vc[4*j+3];
    }
    vb[cur ^ 1][n] = acc;
    vf[(size_t)k * 256 + n] = acc;
    splitf(acc, vh[(size_t)k * 256 + n], vl[(size_t)k * 256 + n]);
    cur ^= 1;
    __syncthreads();
  }
}

// ---------------- batched matrix squaring (3 scales) -----------------------
__global__ void k_asquare(const float* __restrict__ src, float* __restrict__ dst) {
  __shared__ float row[256];
  int s = blockIdx.y, n = blockIdx.x, m = threadIdx.x;
  const float* S = src + (size_t)s * 65536;
  row[m] = S[(size_t)n * 256 + m];
  __syncthreads();
  float acc = 0.f;
  for (int j = 0; j < 256; ++j) acc += row[j] * S[(size_t)j * 256 + m];
  dst[(size_t)s * 65536 + (size_t)n * 256 + m] = acc;
}

// ---------------- V rows [64j..64j+63] = A^64 @ V rows [64(j-1)..] ---------
__global__ void k_vblock(const float* __restrict__ A64, float* __restrict__ VF,
                         unsigned short* __restrict__ VH, unsigned short* __restrict__ VL,
                         int j) {
  const int Ls[3]   = {192, 384, 768};
  const int Voff[3] = {0, 192, 576};
  int s = blockIdx.y;
  if (64 * j >= Ls[s]) return;
  int kl = blockIdx.x, n = threadIdx.x;
  float* vf = VF + (size_t)Voff[s] * 256;
  __shared__ float vold[256];
  vold[n] = vf[(size_t)(64 * (j - 1) + kl) * 256 + n];
  __syncthreads();
  const float4* a4 = (const float4*)(A64 + (size_t)s * 65536 + (size_t)n * 256);
  float acc = 0.f;
  for (int q = 0; q < 64; ++q) {
    float4 a = a4[q];
    acc += a.x * vold[4*q] + a.y * vold[4*q+1] + a.z * vold[4*q+2] + a.w * vold[4*q+3];
  }
  size_t ro = (size_t)(64 * j + kl) * 256 + n;
  vf[ro] = acc;
  splitf(acc, (VH + (size_t)Voff[s] * 256)[ro], (VL + (size_t)Voff[s] * 256)[ro]);
}

// ---------------- F[bj][tau][mc] = E[tau,m] * prefixDFT (hi/lo bf16 pairs) -
__global__ void k_fscan(const float* __restrict__ xn, const float* __restrict__ E,
                        unsigned int* __restrict__ Fh, unsigned int* __restrict__ Fl,
                        int L) {
  int g = blockIdx.x * 256 + threadIdx.x;
  int bj = g >> 6, m = g & 63;
  const float* xr = xn + (size_t)bj * SEQ + (SEQ - L);
  float Sre = 0.f, Sim = 0.f;
  for (int d = 0; d < L; ++d) {
    float x = xr[d];
    float er = E[((size_t)d * 64 + m) * 2], ei = E[((size_t)d * 64 + m) * 2 + 1];
    Sre += x * er; Sim += x * ei;
    int tau = L - 1 - d;
    float fr = E[((size_t)tau * 64 + m) * 2], fi = E[((size_t)tau * 64 + m) * 2 + 1];
    float Fre = fr * Sre - fi * Sim;
    float Fim = fr * Sim + fi * Sre;
    unsigned short hr, lr, hi_, li_;
    splitf(Fre, hr, lr);
    splitf(Fim, hi_, li_);
    size_t idx = ((size_t)bj * L + tau) * 64 + m;
    Fh[idx] = (unsigned)hr | ((unsigned)hi_ << 16);
    Fl[idx] = (unsigned)lr | ((unsigned)li_ << 16);
  }
}

// ---------------- fold spectral weights with irfft@t=191 coefficients ------
__global__ void k_wfold(const float* __restrict__ wre, const float* __restrict__ wim,
                        unsigned short* __restrict__ Wh, unsigned short* __restrict__ Wl,
                        int L) {
  int m = blockIdx.x, i = blockIdx.y, o = threadIdx.x;
  float invL = 1.0f / (float)L, alpha, beta;
  if (m == 0) { alpha = invL; beta = 0.0f; }
  else {
    int r = (m * (PRED - 1)) % L;
    float phi = 6.2831853071795864769f * (float)r / (float)L;
    alpha =  2.0f * invL * cosf(phi);
    beta  = -2.0f * invL * sinf(phi);
  }
  size_t idx = ((size_t)i * 256 + o) * 64 + m;
  float re = wre[idx], im = wim[idx];
  float a = alpha * re + beta * im;
  float b = beta * re - alpha * im;
  size_t ka = ((size_t)i * 128 + 2 * m) * 256 + o;
  size_t kb = ((size_t)i * 128 + 2 * m + 1) * 256 + o;
  splitf(a, Wh[ka], Wl[ka]);
  splitf(b, Wh[kb], Wl[kb]);
}

// ---------------- batched GEMM: xf[bj] = Vmat^T (256xL) @ F_bj (Lx128) -----
// split-bf16: acc += Ahi*Bhi + Alo*Bhi + Ahi*Blo
// lds[0]=Ahi, lds[1]=Alo, lds[2]=Bhi, lds[3]=Blo (single base for imm offsets)
__global__ void __launch_bounds__(256, 1)
k_gemm_xf(const unsigned short* __restrict__ Vh,
          const unsigned short* __restrict__ Vl,
          const unsigned short* __restrict__ Fh,
          const unsigned short* __restrict__ Fl,
          unsigned short* __restrict__ Xh,
          unsigned short* __restrict__ Xl, int L) {
  __shared__ alignas(16) bf16 lds[4][128][LDST];
  int bj = blockIdx.y, iBase = blockIdx.x * 128;
  int tid = threadIdx.x, wave = tid >> 5;
  unsigned lane = lane_id();
  int m = lane & 15, h = lane >> 4;
  int wm = (wave & 3) * 32, wn = (wave >> 2) * 64;
  const bf16* VhB = (const bf16*)Vh;
  const bf16* VlB = (const bf16*)Vl;
  const bf16* FhB = (const bf16*)Fh + (size_t)bj * L * 128;
  const bf16* FlB = (const bf16*)Fl + (size_t)bj * L * 128;
  bf16* XhB = (bf16*)Xh;
  bf16* XlB = (bf16*)Xl;
  v8f acc[2][4] = {};
  int gq = (tid & 31) * 4, gc = tid >> 5;   // quad along contiguous dim, col 0..7
  for (int k0 = 0; k0 < L; k0 += 32) {
    for (int p = 0; p < 4; ++p) {
      int c = gc + 8 * p;
      v4bf a0 = *(const v4bf*)&VhB[(size_t)(k0 + c) * 256 + iBase + gq];
      v4bf a1 = *(const v4bf*)&VlB[(size_t)(k0 + c) * 256 + iBase + gq];
      v4bf b0 = *(const v4bf*)&FhB[(size_t)(k0 + c) * 128 + gq];
      v4bf b1 = *(const v4bf*)&FlB[(size_t)(k0 + c) * 128 + gq];
      for (int q = 0; q < 4; ++q) {
        lds[0][gq + q][c] = a0[q]; lds[1][gq + q][c] = a1[q];
        lds[2][gq + q][c] = b0[q]; lds[3][gq + q][c] = b1[q];
      }
    }
    __syncthreads();
    v16bf ah[2], al[2], bh[4], bl[4];
    for (int t = 0; t < 2; ++t) ah[t] = frag_load(&lds[0][wm + t * 16 + m][0], h);
    for (int t = 0; t < 4; ++t) bh[t] = frag_load(&lds[2][wn + t * 16 + m][0], h);
    for (int a = 0; a < 2; ++a)
      for (int bq = 0; bq < 4; ++bq)
        acc[a][bq] = __builtin_amdgcn_wmma_f32_16x16x32_bf16(
            false, ah[a], false, bh[bq], (short)0, acc[a][bq], false, false);
    for (int t = 0; t < 2; ++t) al[t] = frag_load(&lds[1][wm + t * 16 + m][0], h);
    for (int a = 0; a < 2; ++a)
      for (int bq = 0; bq < 4; ++bq)
        acc[a][bq] = __builtin_amdgcn_wmma_f32_16x16x32_bf16(
            false, al[a], false, bh[bq], (short)0, acc[a][bq], false, false);
    for (int t = 0; t < 4; ++t) bl[t] = frag_load(&lds[3][wn + t * 16 + m][0], h);
    for (int a = 0; a < 2; ++a)
      for (int bq = 0; bq < 4; ++bq)
        acc[a][bq] = __builtin_amdgcn_wmma_f32_16x16x32_bf16(
            false, ah[a], false, bl[bq], (short)0, acc[a][bq], false, false);
    __syncthreads();
  }
  for (int a = 0; a < 2; ++a)
    for (int bq = 0; bq < 4; ++bq)
      for (int r = 0; r < 8; ++r) {
        int gi  = iBase + wm + a * 16 + r + 8 * h;
        int gmc = wn + bq * 16 + m;
        size_t idx = (size_t)bj * KTOT + (size_t)gi * 128 + gmc;
        float v = acc[a][bq][r];
        bf16 hi = (bf16)v;
        XhB[idx] = hi;
        XlB[idx] = (bf16)(v - (float)hi);
      }
}

// ---------------- out1 (256x256) = X~ (256x32768) @ W~ (32768x256), splitK -
__global__ void __launch_bounds__(256, 1)
k_gemm_out(const unsigned short* __restrict__ Xh,
           const unsigned short* __restrict__ Xl,
           const unsigned short* __restrict__ Wh,
           const unsigned short* __restrict__ Wl,
           float* __restrict__ out1) {
  __shared__ alignas(16) bf16 lds[4][128][LDST];
  int bjBase = blockIdx.x * 128, oBase = blockIdx.y * 128;
  int kStart = blockIdx.z * 2048;
  int tid = threadIdx.x, wave = tid >> 5;
  unsigned lane = lane_id();
  int m = lane & 15, h = lane >> 4;
  int wm = (wave & 3) * 32, wn = (wave >> 2) * 64;
  const bf16* XhB = (const bf16*)Xh;
  const bf16* XlB = (const bf16*)Xl;
  const bf16* WhB = (const bf16*)Wh;
  const bf16* WlB = (const bf16*)Wl;
  v8f acc[2][4] = {};
  int akq = tid & 7, arow = tid >> 3;      // A: 8B loads along contiguous k
  int gq = (tid & 31) * 4, gc = tid >> 5;  // B: quad along contiguous o
  for (int k0 = kStart; k0 < kStart + 2048; k0 += 32) {
    for (int rr = 0; rr < 4; ++rr) {
      int row = arow + 32 * rr;
      size_t src = (size_t)(bjBase + row) * KTOT + k0 + 4 * akq;
      *(v4bf*)&lds[0][row][4 * akq] = *(const v4bf*)&XhB[src];
      *(v4bf*)&lds[1][row][4 * akq] = *(const v4bf*)&XlB[src];
    }
    for (int p = 0; p < 4; ++p) {
      int c = gc + 8 * p;
      v4bf b0 = *(const v4bf*)&WhB[(size_t)(k0 + c) * 256 + oBase + gq];
      v4bf b1 = *(const v4bf*)&WlB[(size_t)(k0 + c) * 256 + oBase + gq];
      for (int q = 0; q < 4; ++q) { lds[2][gq + q][c] = b0[q]; lds[3][gq + q][c] = b1[q]; }
    }
    __syncthreads();
    v16bf ah[2], al[2], bh[4], bl[4];
    for (int t = 0; t < 2; ++t) ah[t] = frag_load(&lds[0][wm + t * 16 + m][0], h);
    for (int t = 0; t < 4; ++t) bh[t] = frag_load(&lds[2][wn + t * 16 + m][0], h);
    for (int a = 0; a < 2; ++a)
      for (int bq = 0; bq < 4; ++bq)
        acc[a][bq] = __builtin_amdgcn_wmma_f32_16x16x32_bf16(
            false, ah[a], false, bh[bq], (short)0, acc[a][bq], false, false);
    for (int t = 0; t < 2; ++t) al[t] = frag_load(&lds[1][wm + t * 16 + m][0], h);
    for (int a = 0; a < 2; ++a)
      for (int bq = 0; bq < 4; ++bq)
        acc[a][bq] = __builtin_amdgcn_wmma_f32_16x16x32_bf16(
            false, al[a], false, bh[bq], (short)0, acc[a][bq], false, false);
    for (int t = 0; t < 4; ++t) bl[t] = frag_load(&lds[3][wn + t * 16 + m][0], h);
    for (int a = 0; a < 2; ++a)
      for (int bq = 0; bq < 4; ++bq)
        acc[a][bq] = __builtin_amdgcn_wmma_f32_16x16x32_bf16(
            false, ah[a], false, bl[bq], (short)0, acc[a][bq], false, false);
    __syncthreads();
  }
  for (int a = 0; a < 2; ++a)
    for (int bq = 0; bq < 4; ++bq)
      for (int r = 0; r < 8; ++r) {
        int gbj = bjBase + wm + a * 16 + r + 8 * h;
        int go  = oBase + wn + bq * 16 + m;
        atomicAdd(&out1[(size_t)gbj * 256 + go], acc[a][bq][r]);
      }
}

__global__ void k_zero(float* p, int n) {
  int i = blockIdx.x * 256 + threadIdx.x;
  if (i < n) p[i] = 0.f;
}

// ---------------- legendre projection + scale mix + denorm -----------------
__global__ void k_final(const float* __restrict__ out1, const float* __restrict__ e0,
                        const float* __restrict__ e1, const float* __restrict__ e2,
                        const float* __restrict__ meanstd, const float* __restrict__ aw,
                        const float* __restrict__ ab, const float* __restrict__ mw,
                        const float* __restrict__ mb, float* __restrict__ out) {
  int g = blockIdx.x * 256 + threadIdx.x;   // 0..49151
  int bj = g / PRED, tp = g % PRED;
  int b = bj >> 6, c = bj & 63;
  const float* evs[3] = {e0, e1, e2};
  const int start[3] = {0, 192, 576};
  float xd = mb[0];
  for (int s = 0; s < 3; ++s) {
    const float* ev = evs[s] + (size_t)(start[s] + tp) * 256;
    const float* o1 = out1 + (size_t)s * 65536 + (size_t)bj * 256;
    float acc = 0.f;
    for (int o = 0; o < 256; ++o) acc += o1[o] * ev[o];
    xd += acc * mw[s];
  }
  float mean = meanstd[bj], sd = meanstd[256 + bj];
  out[((size_t)b * PRED + tp) * 64 + c] = (xd - ab[c]) / (aw[c] + 1e-10f) * sd + mean;
}

extern "C" void kernel_launch(void* const* d_in, const int* in_sizes, int n_in,
                              void* d_out, int out_size, void* d_ws, size_t ws_size,
                              hipStream_t stream) {
  (void)in_sizes; (void)n_in; (void)out_size; (void)ws_size;
  const float* x_enc = (const float*)d_in[0];
  const float* aw    = (const float*)d_in[4];
  const float* ab    = (const float*)d_in[5];
  const float* w1re  = (const float*)d_in[6];
  const float* w1im  = (const float*)d_in[7];
  const float* mw    = (const float*)d_in[8];
  const float* mb    = (const float*)d_in[9];
  const float* Amats = (const float*)d_in[10];
  const float* Bvecs = (const float*)d_in[11];
  const float* e0    = (const float*)d_in[12];
  const float* e1    = (const float*)d_in[13];
  const float* e2    = (const float*)d_in[14];

  char* ws = (char*)d_ws;
  size_t off = 0;
  auto alloc = [&](size_t bytes) -> void* {
    void* p = ws + off;
    off = (off + bytes + 255) & ~(size_t)255;
    return p;
  };
  float* xn            = (float*)alloc((size_t)256 * 768 * 4);
  float* meanstd       = (float*)alloc(512 * 4);
  float* Etab          = (float*)alloc((size_t)1344 * 64 * 2 * 4);
  float* VF            = (float*)alloc((size_t)1344 * 256 * 4);
  unsigned short* VBh  = (unsigned short*)alloc((size_t)1344 * 256 * 2);
  unsigned short* VBl  = (unsigned short*)alloc((size_t)1344 * 256 * 2);
  float* Apow0         = (float*)alloc((size_t)3 * 65536 * 4);
  float* Apow1         = (float*)alloc((size_t)3 * 65536 * 4);
  unsigned int* Fh     = (unsigned int*)alloc((size_t)256 * 768 * 64 * 4);
  unsigned int* Fl     = (unsigned int*)alloc((size_t)256 * 768 * 64 * 4);
  unsigned short* Xh   = (unsigned short*)alloc((size_t)256 * 32768 * 2);
  unsigned short* Xl   = (unsigned short*)alloc((size_t)256 * 32768 * 2);
  unsigned short* Wh   = (unsigned short*)alloc((size_t)32768 * 256 * 2);
  unsigned short* Wl   = (unsigned short*)alloc((size_t)32768 * 256 * 2);
  float* out1          = (float*)alloc((size_t)3 * 65536 * 4);

  k_normalize<<<256, 128, 0, stream>>>(x_enc, aw, ab, xn, meanstd);
  k_etab<<<dim3(192, 3), 256, 0, stream>>>(Etab);
  k_vseed<<<3, 256, 0, stream>>>(Amats, Bvecs, VF, VBh, VBl);
  k_asquare<<<dim3(256, 3), 256, 0, stream>>>(Amats, Apow1);   // A^2
  k_asquare<<<dim3(256, 3), 256, 0, stream>>>(Apow1, Apow0);   // A^4
  k_asquare<<<dim3(256, 3), 256, 0, stream>>>(Apow0, Apow1);   // A^8
  k_asquare<<<dim3(256, 3), 256, 0, stream>>>(Apow1, Apow0);   // A^16
  k_asquare<<<dim3(256, 3), 256, 0, stream>>>(Apow0, Apow1);   // A^32
  k_asquare<<<dim3(256, 3), 256, 0, stream>>>(Apow1, Apow0);   // A^64
  for (int j = 1; j <= 11; ++j)
    k_vblock<<<dim3(64, 3), 256, 0, stream>>>(Apow0, VF, VBh, VBl, j);
  k_zero<<<768, 256, 0, stream>>>(out1, 3 * 65536);

  const int Ls[3]   = {192, 384, 768};
  const int Eoff[3] = {0, 24576, 73728};
  const int Voff[3] = {0, 192, 576};
  for (int s = 0; s < 3; ++s) {
    int L = Ls[s];
    k_fscan<<<64, 256, 0, stream>>>(xn, Etab + Eoff[s], Fh, Fl, L);
    k_wfold<<<dim3(64, 256), 256, 0, stream>>>(
        w1re + (size_t)s * 256 * 256 * 64, w1im + (size_t)s * 256 * 256 * 64,
        Wh, Wl, L);
    k_gemm_xf<<<dim3(2, 256), 256, 0, stream>>>(
        VBh + (size_t)Voff[s] * 256, VBl + (size_t)Voff[s] * 256,
        (const unsigned short*)Fh, (const unsigned short*)Fl, Xh, Xl, L);
    k_gemm_out<<<dim3(2, 2, 16), 256, 0, stream>>>(Xh, Xl, Wh, Wl,
                                                   out1 + (size_t)s * 65536);
  }
  k_final<<<192, 256, 0, stream>>>(out1, e0, e1, e2, meanstd, aw, ab, mw, mb,
                                   (float*)d_out);
}
